// NetGAT_27178553049851
// MI455X (gfx1250) — matrix operand order repaired and checked by reading, not compiled, not run on hardware
//
#include <hip/hip_runtime.h>
#include <cstdint>
#include <cfloat>

// ---------------------------------------------------------------------------
// Problem constants (match reference)
// ---------------------------------------------------------------------------
#define NUM_GRAPHS 64
#define NODES_PER_GRAPH 512
#define NN (NUM_GRAPHS * NODES_PER_GRAPH)   // 32768 nodes
#define KNB 16                              // neighbours per node
#define HID 256                             // 8 heads * 32 dims
#define BN_INV 0.9999950000374997f          // 1/sqrt(1 + 1e-5)

typedef __attribute__((ext_vector_type(16))) _Float16 v16h;
typedef __attribute__((ext_vector_type(8)))  float    v8f;

union Frag { v16h v; _Float16 h[16]; uint32_t u[8]; };

// A-operand K offset for (vgpr v, hi half): v0..3 -> K=2v (+8 hi), v4..7 -> 16+2(v-4) (+8 hi)
__device__ __forceinline__ int a_koff(int v, int hi) {
  return ((v & 4) ? (16 + 2 * (v & 3)) : (2 * v)) + (hi << 3);
}

__device__ __forceinline__ uint32_t pack2h(float a, float b) {
  union { _Float16 h[2]; uint32_t u; } p;
  p.h[0] = (_Float16)a; p.h[1] = (_Float16)b;
  return p.u;
}

__device__ __forceinline__ v8f wmma16(const Frag& a, const Frag& b, v8f c) {
  return __builtin_amdgcn_wmma_f32_16x16x32_f16(false, a.v, false, b.v, (short)0, c,
                                                false, false);
}

// Async 16-byte global -> LDS copy (per lane). Generic LDS pointers carry the
// LDS byte address in their low 32 bits (ISA aperture rule), which is what the
// VDST VGPR of global_load_async_to_lds_* expects. Tracked by ASYNCcnt.
__device__ __forceinline__ void async_g2l_b128(void* lds_dst, const void* gsrc) {
  uint32_t ldsa = (uint32_t)(uintptr_t)lds_dst;
  uint64_t ga   = (uint64_t)(uintptr_t)gsrc;
  asm volatile("global_load_async_to_lds_b128 %0, %1, off"
               :: "v"(ldsa), "v"(ga) : "memory");
}
__device__ __forceinline__ void wait_asynccnt0() {
  asm volatile("s_wait_asynccnt 0" ::: "memory");
}

// ---------------------------------------------------------------------------
// Pre-swizzle an f32 weight matrix W[K, ncols] (row-major) into packed f16
// B-operand tiles: out[((tile_off+tl)*ksteps+s)*8+v)*32+lane].
// B layout (32x16 f16): lane%16 = column, vgpr v holds K=a_koff(v,hi),+1
// (same K order as the A operand); out-of-range K rows packed as 0 (K padding).
// ---------------------------------------------------------------------------
__global__ void pack_b_kernel(const float* __restrict__ W, int K, int ncols,
                              uint32_t* __restrict__ out, int ksteps, int tile_offset) {
  int total = (ncols >> 4) * ksteps * 256;
  int tid = blockIdx.x * blockDim.x + threadIdx.x;
  if (tid >= total) return;
  int lane = tid & 31;
  int v    = (tid >> 5) & 7;
  int rest = tid >> 8;
  int s    = rest % ksteps;
  int tl   = rest / ksteps;
  int hi   = lane >> 4;
  int col  = tl * 16 + (lane & 15);
  int k0   = s * 32 + a_koff(v, hi);
  float v0 = (k0     < K) ? W[k0 * ncols + col]       : 0.f;
  float v1 = (k0 + 1 < K) ? W[(k0 + 1) * ncols + col] : 0.f;
  out[(((tile_offset + tl) * ksteps + s) * 8 + v) * 32 + lane] = pack2h(v0, v1);
}

// ---------------------------------------------------------------------------
// EdgeConv: one wave per node. The node's 16 incoming edges are the M=16 rows.
// m = [x_i, x_j - x_i] (K padded 6->32) -> 3x (Lin,BN,ReLU) via WMMA ->
// column-wise max over rows -> h64[node, 64] (f16).
// ---------------------------------------------------------------------------
__global__ void __launch_bounds__(256) edgeconv_kernel(
    const float* __restrict__ x, const int* __restrict__ src,
    const float* __restrict__ b1, const float* __restrict__ g1, const float* __restrict__ be1,
    const float* __restrict__ b2, const float* __restrict__ g2, const float* __restrict__ be2,
    const float* __restrict__ b3, const float* __restrict__ g3, const float* __restrict__ be3,
    const uint32_t* __restrict__ pw, _Float16* __restrict__ h64) {
  __shared__ _Float16 lds[8][16][64];           // per-wave activation tile
  const int lane = threadIdx.x & 31;
  const int wid  = threadIdx.x >> 5;
  const int node = blockIdx.x * 8 + wid;
  const int hi   = lane >> 4;
  const int r    = lane & 15;                   // row (edge) index

  float xi0 = x[node * 3 + 0], xi1 = x[node * 3 + 1], xi2 = x[node * 3 + 2];
  int sidx = src[node * KNB + r];
  float m0 = xi0, m1 = xi1, m2 = xi2;
  float m3 = x[sidx * 3 + 0] - xi0;
  float m4 = x[sidx * 3 + 1] - xi1;
  float m5 = x[sidx * 3 + 2] - xi2;

  // ---- layer 1: A = 16x32 (only K=0..5 nonzero, held by lo lanes) ----
  Frag a;
  #pragma unroll
  for (int i = 0; i < 8; ++i) a.u[i] = 0;
  if (!hi) {
    a.u[0] = pack2h(m0, m1);   // K=0,1
    a.u[1] = pack2h(m2, m3);   // K=2,3
    a.u[2] = pack2h(m4, m5);   // K=4,5
  }
  #pragma unroll
  for (int t = 0; t < 4; ++t) {
    Frag b;
    #pragma unroll
    for (int v = 0; v < 8; ++v) b.u[v] = pw[(t * 8 + v) * 32 + lane];
    v8f c = {0, 0, 0, 0, 0, 0, 0, 0};
    c = wmma16(a, b, c);
    int col = t * 16 + r;
    float bias = b1[col], gg = g1[col] * BN_INV, bb = be1[col];
    #pragma unroll
    for (int v = 0; v < 8; ++v) {
      float val = fmaxf((c[v] + bias) * gg + bb, 0.f);
      lds[wid][v + hi * 8][col] = (_Float16)val;
    }
  }
  asm volatile("" ::: "memory");   // keep LDS store->load program order (in-order per wave)

  // ---- layer 2: K=64 (2 ksteps) ----
  const uint32_t* pw2 = pw + 1024;
  Frag a2[2];
  #pragma unroll
  for (int s2 = 0; s2 < 2; ++s2)
    #pragma unroll
    for (int v = 0; v < 8; ++v) {
      int k0 = s2 * 32 + a_koff(v, hi);
      a2[s2].u[v] = *(const uint32_t*)&lds[wid][r][k0];
    }
  float out2[4][8];
  #pragma unroll
  for (int t = 0; t < 4; ++t) {
    v8f c = {0, 0, 0, 0, 0, 0, 0, 0};
    #pragma unroll
    for (int s2 = 0; s2 < 2; ++s2) {
      Frag b;
      #pragma unroll
      for (int v = 0; v < 8; ++v) b.u[v] = pw2[((t * 2 + s2) * 8 + v) * 32 + lane];
      c = wmma16(a2[s2], b, c);
    }
    int col = t * 16 + r;
    float bias = b2[col], gg = g2[col] * BN_INV, bb = be2[col];
    #pragma unroll
    for (int v = 0; v < 8; ++v) out2[t][v] = fmaxf((c[v] + bias) * gg + bb, 0.f);
  }
  #pragma unroll
  for (int t = 0; t < 4; ++t) {
    int col = t * 16 + r;
    #pragma unroll
    for (int v = 0; v < 8; ++v) lds[wid][v + hi * 8][col] = (_Float16)out2[t][v];
  }
  asm volatile("" ::: "memory");

  // ---- layer 3: K=64, then max over the 16 rows (edges) ----
  const uint32_t* pw3 = pw + 3072;
  Frag a3[2];
  #pragma unroll
  for (int s2 = 0; s2 < 2; ++s2)
    #pragma unroll
    for (int v = 0; v < 8; ++v) {
      int k0 = s2 * 32 + a_koff(v, hi);
      a3[s2].u[v] = *(const uint32_t*)&lds[wid][r][k0];
    }
  #pragma unroll
  for (int t = 0; t < 4; ++t) {
    v8f c = {0, 0, 0, 0, 0, 0, 0, 0};
    #pragma unroll
    for (int s2 = 0; s2 < 2; ++s2) {
      Frag b;
      #pragma unroll
      for (int v = 0; v < 8; ++v) b.u[v] = pw3[((t * 2 + s2) * 8 + v) * 32 + lane];
      c = wmma16(a3[s2], b, c);
    }
    int col = t * 16 + r;
    float bias = b3[col], gg = g3[col] * BN_INV, bb = be3[col];
    float mx = -FLT_MAX;
    #pragma unroll
    for (int v = 0; v < 8; ++v)
      mx = fmaxf(mx, fmaxf((c[v] + bias) * gg + bb, 0.f));
    mx = fmaxf(mx, __shfl_xor(mx, 16, 32));   // combine rows held by hi/lo lane halves
    if (!hi) h64[node * 64 + col] = (_Float16)mx;
  }
}

// ---------------------------------------------------------------------------
// hw = h @ [W | Wl]  : WMMA GEMM. One block per 16-row stripe; the stripe of
// A (16 x fin f16, contiguous in memory) is async-copied into LDS once and
// shared by all 8 waves (each wave covers 64 of the 512 output columns).
// B fragments are pre-packed in global; next K-step's B is prefetched.
// ---------------------------------------------------------------------------
__global__ void __launch_bounds__(256) gat_gemm_kernel(
    const _Float16* __restrict__ A, const uint32_t* __restrict__ bp,
    float* __restrict__ C, int fin) {
  __shared__ _Float16 sA[16 * 256];            // up to 8 KB stripe
  const int tid   = threadIdx.x;
  const int lane  = tid & 31;
  const int ng    = tid >> 5;                  // wave id = column group
  const int mtile = blockIdx.x;
  const int hi    = lane >> 4;
  const int r     = lane & 15;

  // stage A stripe: rows mtile*16 .. +15 are one contiguous 16*fin*2 B block
  {
    const char* gsrc = (const char*)(A + (size_t)mtile * 16 * fin);
    char*       ldst = (char*)(void*)sA;
    int nbytes = 16 * fin * 2;
    for (int off = tid * 16; off < nbytes; off += 256 * 16)
      async_g2l_b128(ldst + off, gsrc + off);
    wait_asynccnt0();
  }
  __syncthreads();

  const int ks = fin >> 5;
  v8f acc[4];
  #pragma unroll
  for (int t = 0; t < 4; ++t) acc[t] = (v8f){0, 0, 0, 0, 0, 0, 0, 0};

  for (int s = 0; s < ks; ++s) {
    Frag a;
    #pragma unroll
    for (int v = 0; v < 8; ++v)
      a.u[v] = *(const uint32_t*)&sA[r * fin + s * 32 + a_koff(v, hi)];
    if (s + 1 < ks)   // prefetch next K-step's B fragments for this wave
      __builtin_prefetch(&bp[(((ng * 4) * ks + s + 1) * 8) * 32 + lane], 0, 0);
    #pragma unroll
    for (int t = 0; t < 4; ++t) {
      int tile = ng * 4 + t;
      Frag b;
      int base = ((tile * ks + s) * 8) * 32 + lane;
      #pragma unroll
      for (int v = 0; v < 8; ++v) b.u[v] = bp[base + v * 32];
      acc[t] = wmma16(a, b, acc[t]);
    }
  }
  #pragma unroll
  for (int t = 0; t < 4; ++t) {
    int col = (ng * 4 + t) * 16 + r;
    #pragma unroll
    for (int v = 0; v < 8; ++v)
      C[(size_t)(mtile * 16 + v + hi * 8) * 512 + col] = acc[t][v];
  }
}

// ---------------------------------------------------------------------------
// Attention coefficients: asrc/adst[n,h] = sum_d h'(n,h,d) * a_src/dst(h,d)
// ---------------------------------------------------------------------------
__global__ void att_coeff_kernel(const float* __restrict__ hw,
                                 const float* __restrict__ a_src,
                                 const float* __restrict__ a_dst,
                                 float* __restrict__ asrc, float* __restrict__ adst) {
  int t = blockIdx.x * 256 + threadIdx.x;
  int node = t >> 3, hd = t & 7;
  if (node >= NN) return;
  const float* h  = hw + (size_t)node * 512 + hd * 32;
  const float* as = a_src + hd * 32;
  const float* ad = a_dst + hd * 32;
  float s1 = 0.f, s2 = 0.f;
  #pragma unroll
  for (int d = 0; d < 32; ++d) { s1 += h[d] * as[d]; s2 += h[d] * ad[d]; }
  asrc[node * 8 + hd] = s1;
  adst[node * 8 + hd] = s2;
}

// ---------------------------------------------------------------------------
// GAT aggregation: one block per node. 17 edges (16 neighbours + self-loop).
// leaky_relu(0.2) -> per-head softmax (+1e-16) -> weighted sum of h'[src].
// xi = agg + gat_b + (h@Wl part of hw) + lin_b. Writes xi (f32) and relu(xi) (f16).
// ---------------------------------------------------------------------------
__global__ void __launch_bounds__(256) gat_agg_kernel(
    const float* __restrict__ hw, const int* __restrict__ src,
    const float* __restrict__ asrc, const float* __restrict__ adst,
    const float* __restrict__ gat_b, const float* __restrict__ lin_b,
    float* __restrict__ xi_out, _Float16* __restrict__ h256) {
  __shared__ int   sidx[17];
  __shared__ float ev[17][8];
  __shared__ float alpha[17][8];
  int node = blockIdx.x;
  int t = threadIdx.x;
  if (t < 16) sidx[t] = src[node * KNB + t];
  if (t == 16) sidx[16] = node;                 // self loop
  __syncthreads();
  if (t < 136) {
    int e = t >> 3, hd = t & 7;
    float v = asrc[sidx[e] * 8 + hd] + adst[node * 8 + hd];
    ev[e][hd] = (v >= 0.f) ? v : 0.2f * v;
  }
  __syncthreads();
  if (t < 8) {
    float m = -FLT_MAX;
    for (int e = 0; e < 17; ++e) m = fmaxf(m, ev[e][t]);
    float s = 0.f;
    for (int e = 0; e < 17; ++e) s += __expf(ev[e][t] - m);
    float inv = 1.f / (s + 1e-16f);
    for (int e = 0; e < 17; ++e) alpha[e][t] = __expf(ev[e][t] - m) * inv;
  }
  __syncthreads();
  int c = t, hd = c >> 5;
  float acc = 0.f;
  #pragma unroll 1
  for (int e = 0; e < 17; ++e) acc += alpha[e][hd] * hw[(size_t)sidx[e] * 512 + c];
  float xi = acc + gat_b[c] + hw[(size_t)node * 512 + 256 + c] + lin_b[c];
  xi_out[(size_t)node * 256 + c] = xi;
  h256[(size_t)node * 256 + c] = (_Float16)fmaxf(xi, 0.f);
}

// ---------------------------------------------------------------------------
// Per-graph max + mean pooling of this layer's xi into f[64, 2048].
// ---------------------------------------------------------------------------
__global__ void pool_kernel(const float* __restrict__ xi, float* __restrict__ fbuf,
                            int layer) {
  int g = blockIdx.x, c = threadIdx.x;
  float mx = -FLT_MAX, sm = 0.f;
  for (int n = 0; n < NODES_PER_GRAPH; ++n) {
    float v = xi[(size_t)(g * NODES_PER_GRAPH + n) * 256 + c];
    mx = fmaxf(mx, v);
    sm += v;
  }
  fbuf[g * 2048 + layer * 256 + c]        = mx;
  fbuf[g * 2048 + 1024 + layer * 256 + c] = sm * (1.f / NODES_PER_GRAPH);
}

// ---------------------------------------------------------------------------
// Head linear layers (64 rows only -> plain f32 dot products are fine)
// ---------------------------------------------------------------------------
__global__ void dense_kernel(const float* __restrict__ in, const float* __restrict__ W,
                             const float* __restrict__ b, float* __restrict__ out,
                             int fin, int fout) {
  int g = blockIdx.x, j = threadIdx.x;
  if (j >= fout) return;
  float acc = b[j];
  const float* row = in + (size_t)g * fin;
  for (int k = 0; k < fin; ++k) acc += row[k] * W[(size_t)k * fout + j];
  out[(size_t)g * fout + j] = acc;
}

__global__ void head3_kernel(const float* __restrict__ in, const float* __restrict__ W,
                             const float* __restrict__ b, float* __restrict__ out) {
  __shared__ float y[40];
  __shared__ float red[2];
  int g = blockIdx.x, j = threadIdx.x;
  if (j < 40) {
    float acc = b[j];
    for (int k = 0; k < 256; ++k) acc += in[g * 256 + k] * W[k * 40 + j];
    y[j] = acc;
  }
  __syncthreads();
  if (j == 0) {
    float m = -FLT_MAX;
    for (int i = 0; i < 40; ++i) m = fmaxf(m, y[i]);
    float s = 0.f;
    for (int i = 0; i < 40; ++i) s += __expf(y[i] - m);
    red[0] = m;
    red[1] = __logf(s);
  }
  __syncthreads();
  if (j < 40) out[g * 40 + j] = y[j] - red[0] - red[1];
}

// ---------------------------------------------------------------------------
// Host launcher
// ---------------------------------------------------------------------------
extern "C" void kernel_launch(void* const* d_in, const int* in_sizes, int n_in,
                              void* d_out, int out_size, void* d_ws, size_t ws_size,
                              hipStream_t stream) {
  const float* x    = (const float*)d_in[0];
  const int*   edge = (const int*)d_in[1];        // row 0 = src (first E entries)

  const float *ecW[3], *ecB[3], *ecG[3], *ecBE[3];
  for (int i = 0; i < 3; ++i) {
    ecW[i]  = (const float*)d_in[3 + 4 * i];
    ecB[i]  = (const float*)d_in[4 + 4 * i];
    ecG[i]  = (const float*)d_in[5 + 4 * i];
    ecBE[i] = (const float*)d_in[6 + 4 * i];
  }
  const float *gW[4], *gAs[4], *gAd[4], *gB[4], *lW[4], *lB[4];
  for (int l = 0; l < 4; ++l) {
    gW[l]  = (const float*)d_in[15 + 4 * l];
    gAs[l] = (const float*)d_in[16 + 4 * l];
    gAd[l] = (const float*)d_in[17 + 4 * l];
    gB[l]  = (const float*)d_in[18 + 4 * l];
    lW[l]  = (const float*)d_in[31 + 2 * l];
    lB[l]  = (const float*)d_in[32 + 2 * l];
  }
  const float* W1 = (const float*)d_in[39];
  const float* b1 = (const float*)d_in[40];
  const float* W2 = (const float*)d_in[41];
  const float* b2 = (const float*)d_in[42];
  const float* W3 = (const float*)d_in[43];
  const float* b3 = (const float*)d_in[44];
  float* out = (float*)d_out;

  // workspace layout (bytes)
  uint8_t* ws = (uint8_t*)d_ws;
  uint32_t*  pw     = (uint32_t*)(ws);                              //  <1 MB packed f16 weights
  _Float16*  h64    = (_Float16*)(ws + (size_t)(1)   * (1u << 20)); //  4 MB
  _Float16*  h256   = (_Float16*)(ws + (size_t)(5)   * (1u << 20)); // 16 MB
  float*     hw     = (float*)   (ws + (size_t)(21)  * (1u << 20)); // 64 MB
  float*     asrc   = (float*)   (ws + (size_t)(85)  * (1u << 20)); //  1 MB
  float*     adst   = (float*)   (ws + (size_t)(86)  * (1u << 20)); //  1 MB
  float*     xi_buf = (float*)   (ws + (size_t)(87)  * (1u << 20)); // 32 MB
  float*     fbuf   = (float*)   (ws + (size_t)(119) * (1u << 20)); // 0.5 MB
  float*     y1     = (float*)   (ws + (size_t)(120) * (1u << 20));
  float*     y2     = (float*)   (ws + (size_t)(120) * (1u << 20) + (1u << 19));

  // --- pre-swizzle all weights into WMMA B layout (f16) ---
  auto pack = [&](const float* W, int K, int ncols, uint32_t* dst, int toff) {
    int ks = (K + 31) / 32;
    int total = (ncols / 16) * ks * 256;
    pack_b_kernel<<<(total + 255) / 256, 256, 0, stream>>>(W, K, ncols, dst, ks, toff);
  };
  pack(ecW[0], 6,  64, pw + 0,    0);
  pack(ecW[1], 64, 64, pw + 1024, 0);
  pack(ecW[2], 64, 64, pw + 3072, 0);
  const int gatOff[4] = {5120, 21504, 87040, 152576};
  for (int l = 0; l < 4; ++l) {
    int fin = (l == 0) ? 64 : 256;
    pack(gW[l], fin, 256, pw + gatOff[l], 0);    // cols 0..255   = W
    pack(lW[l], fin, 256, pw + gatOff[l], 16);   // cols 256..511 = Wl
  }

  // --- EdgeConv + max aggregation -> h64 (f16) ---
  edgeconv_kernel<<<NN / 8, 256, 0, stream>>>(
      x, edge, ecB[0], ecG[0], ecBE[0], ecB[1], ecG[1], ecBE[1],
      ecB[2], ecG[2], ecBE[2], pw, h64);

  // --- 4 GAT layers ---
  for (int l = 0; l < 4; ++l) {
    int fin = (l == 0) ? 64 : 256;
    gat_gemm_kernel<<<NN / 16, 256, 0, stream>>>(
        (l == 0) ? h64 : h256, pw + gatOff[l], hw, fin);
    att_coeff_kernel<<<NN * 8 / 256, 256, 0, stream>>>(hw, gAs[l], gAd[l], asrc, adst);
    gat_agg_kernel<<<NN, 256, 0, stream>>>(hw, edge, asrc, adst, gB[l], lB[l],
                                           xi_buf, h256);
    pool_kernel<<<NUM_GRAPHS, 256, 0, stream>>>(xi_buf, fbuf, l);
  }

  // --- head MLP + log_softmax ---
  dense_kernel<<<NUM_GRAPHS, 512, 0, stream>>>(fbuf, W1, b1, y1, 2048, 512);
  dense_kernel<<<NUM_GRAPHS, 256, 0, stream>>>(y1, W2, b2, y2, 512, 256);
  head3_kernel<<<NUM_GRAPHS, 64, 0, stream>>>(y2, W3, b3, out);
}
